// KMeans_76209899700926
// MI455X (gfx1250) — matrix operand order repaired and checked by reading, not compiled, not run on hardware
//
#include <hip/hip_runtime.h>

typedef _Float16 v16h __attribute__((ext_vector_type(16)));
typedef _Float16 v8h  __attribute__((ext_vector_type(8)));
typedef _Float16 v4h  __attribute__((ext_vector_type(4)));
typedef float    v8f  __attribute__((ext_vector_type(8)));

#define DIM 64
#define NCLUST 256
#define PTS_PER_BLOCK 128
#define THREADS 256

union V16 {
    v16h v;
    v8h  h[2];
};

// ---------------------------------------------------------------------------
// Prep: split centroids into f16 hi/lo pairs and compute ||c||^2.
// ---------------------------------------------------------------------------
__global__ void kmeans_prep_centroids(const float* __restrict__ c,
                                      _Float16* __restrict__ chi,
                                      _Float16* __restrict__ clo,
                                      float* __restrict__ csq) {
    int k = blockIdx.x * blockDim.x + threadIdx.x;
    if (k >= NCLUST) return;
    float s = 0.0f;
    for (int d = 0; d < DIM; ++d) {
        float v = c[k * DIM + d];
        s += v * v;
        _Float16 hi = (_Float16)v;
        chi[k * DIM + d] = hi;
        clo[k * DIM + d] = (_Float16)(v - (float)hi);
    }
    csq[k] = s;
}

// ---------------------------------------------------------------------------
// Main: assignments via WMMA f32_16x16x32_f16 with split-f16 (hi/lo) GEMM.
// Rank key is ||c||^2 - 2 x.c  (||x||^2 is row-constant, dropped for argmin).
// ---------------------------------------------------------------------------
__global__ __launch_bounds__(THREADS)
void kmeans_assign_wmma(const float* __restrict__ x,
                        const _Float16* __restrict__ chi,
                        const _Float16* __restrict__ clo,
                        const float* __restrict__ csq,
                        float* __restrict__ out_assign,
                        int n) {
    __shared__ __align__(32) _Float16 sXhi[PTS_PER_BLOCK * DIM];
    __shared__ __align__(32) _Float16 sXlo[PTS_PER_BLOCK * DIM];
    __shared__ __align__(32) _Float16 sChi[NCLUST * DIM];
    __shared__ __align__(32) _Float16 sClo[NCLUST * DIM];
    __shared__ float sCsq[NCLUST];

    const int t    = threadIdx.x;
    const int lane = t & 31;
    const int wave = t >> 5;
    const long long base = (long long)blockIdx.x * PTS_PER_BLOCK;

    // ---- stage centroids (hi/lo) + ||c||^2 into LDS (128-bit copies) ----
    {
        const uint4* g0 = (const uint4*)chi;
        const uint4* g1 = (const uint4*)clo;
        uint4* s0 = (uint4*)sChi;
        uint4* s1 = (uint4*)sClo;
        const int nvec = NCLUST * DIM * (int)sizeof(_Float16) / 16;  // 2048
        for (int i = t; i < nvec; i += THREADS) {
            s0[i] = g0[i];
            s1[i] = g1[i];
        }
        for (int i = t; i < NCLUST; i += THREADS) sCsq[i] = csq[i];
    }

    // ---- stage x tile (float4 global loads), split fp32 -> f16 hi + lo ----
    {
        const float4* xg = (const float4*)x;          // hipMalloc => 16B aligned
        const int CH = DIM / 4;                       // float4 chunks per point
        for (int i = t; i < PTS_PER_BLOCK * CH; i += THREADS) {
            const int p    = i / CH;
            const int col4 = i % CH;
            long long gp = base + p;
            if (gp >= n) gp = n - 1;                  // clamp (writes guarded later)
            float4 v = xg[gp * CH + col4];
            v4h hi, lo;
            hi[0] = (_Float16)v.x; lo[0] = (_Float16)(v.x - (float)hi[0]);
            hi[1] = (_Float16)v.y; lo[1] = (_Float16)(v.y - (float)hi[1]);
            hi[2] = (_Float16)v.z; lo[2] = (_Float16)(v.z - (float)hi[2]);
            hi[3] = (_Float16)v.w; lo[3] = (_Float16)(v.w - (float)hi[3]);
            *(v4h*)&sXhi[i * 4] = hi;                 // 8B-aligned ds_store_b64
            *(v4h*)&sXlo[i * 4] = lo;
        }
    }
    __syncthreads();

    // ---- build persistent A fragments (16 points per wave) ----
    // 16-bit A 16x32 layout: lane -> M = lane&15; lanes>=16 shift K by 8;
    // elements 0..7 -> K=b..b+7, elements 8..15 -> K=b+16..b+23.
    const int m  = lane & 15;
    const int kb = (lane >= 16) ? 8 : 0;
    const _Float16* xrh = &sXhi[(wave * 16 + m) * DIM];
    const _Float16* xrl = &sXlo[(wave * 16 + m) * DIM];
    V16 aHi[2], aLo[2];
#pragma unroll
    for (int f = 0; f < 2; ++f) {
        const int o = kb + 32 * f;
        aHi[f].h[0] = *(const v8h*)(xrh + o);
        aHi[f].h[1] = *(const v8h*)(xrh + o + 16);
        aLo[f].h[0] = *(const v8h*)(xrl + o);
        aLo[f].h[1] = *(const v8h*)(xrl + o + 16);
    }

    // ---- sweep 16 cluster tiles of 16, tracking per-row (min, argmin) ----
    // C/D layout: VGPR r, lanes 0-15 -> M=r, lanes 16-31 -> M=8+r; N = lane&15.
    // B 32x16 layout: N = lane&15; lanes>=16 shift K by 8; elements 0..7 -> K=koff..koff+7, elements 8..15 -> K=koff+16..koff+23.
    const int nb   = lane & 15;
    const int koff = (lane >= 16) ? 8 : 0;
    float minv[8];
    int   mini[8];
#pragma unroll
    for (int r = 0; r < 8; ++r) { minv[r] = 3.4e38f; mini[r] = 0; }

#pragma unroll 4
    for (int tile = 0; tile < NCLUST / 16; ++tile) {
        const _Float16* crh = &sChi[(tile * 16 + nb) * DIM];
        const _Float16* crl = &sClo[(tile * 16 + nb) * DIM];
        V16 bHi[2], bLo[2];
#pragma unroll
        for (int f = 0; f < 2; ++f) {
            const int o = koff + 32 * f;
            bHi[f].h[0] = *(const v8h*)(crh + o);
            bHi[f].h[1] = *(const v8h*)(crh + o + 16);
            bLo[f].h[0] = *(const v8h*)(crl + o);
            bLo[f].h[1] = *(const v8h*)(crl + o + 16);
        }

        v8f acc = {};
        // hi*hi + hi*lo + lo*hi  (lo*lo term ~1e-5, negligible)
        acc = __builtin_amdgcn_wmma_f32_16x16x32_f16(false, aHi[0].v, false, bHi[0].v, (short)0, acc, false, false);
        acc = __builtin_amdgcn_wmma_f32_16x16x32_f16(false, aHi[1].v, false, bHi[1].v, (short)0, acc, false, false);
        acc = __builtin_amdgcn_wmma_f32_16x16x32_f16(false, aHi[0].v, false, bLo[0].v, (short)0, acc, false, false);
        acc = __builtin_amdgcn_wmma_f32_16x16x32_f16(false, aHi[1].v, false, bLo[1].v, (short)0, acc, false, false);
        acc = __builtin_amdgcn_wmma_f32_16x16x32_f16(false, aLo[0].v, false, bHi[0].v, (short)0, acc, false, false);
        acc = __builtin_amdgcn_wmma_f32_16x16x32_f16(false, aLo[1].v, false, bHi[1].v, (short)0, acc, false, false);

        const float cs  = sCsq[tile * 16 + nb];
        const int   idx = tile * 16 + nb;
#pragma unroll
        for (int r = 0; r < 8; ++r) {
            float dkey = cs - 2.0f * acc[r];          // == d^2 - ||x||^2 (row const)
            if (dkey < minv[r]) { minv[r] = dkey; mini[r] = idx; }
        }
    }

    // ---- reduce (min, argmin) across the 16 lanes sharing each row ----
#pragma unroll
    for (int r = 0; r < 8; ++r) {
        float v = minv[r];
        int   i = mini[r];
#pragma unroll
        for (int off = 8; off >= 1; off >>= 1) {      // stays within 16-lane halves
            float ov = __shfl_xor(v, off, 32);
            int   oi = __shfl_xor(i, off, 32);
            if (ov < v || (ov == v && oi < i)) { v = ov; i = oi; }
        }
        minv[r] = v;
        mini[r] = i;
    }

    if (lane == 0 || lane == 16) {
        const int rowbase = wave * 16 + ((lane == 16) ? 8 : 0);
#pragma unroll
        for (int r = 0; r < 8; ++r) {
            long long gp = base + rowbase + r;
            if (gp < n) out_assign[gp] = (float)mini[r];
        }
    }
}

// ---------------------------------------------------------------------------
extern "C" void kernel_launch(void* const* d_in, const int* in_sizes, int n_in,
                              void* d_out, int out_size, void* d_ws, size_t ws_size,
                              hipStream_t stream) {
    const float* x = (const float*)d_in[0];
    const float* c = (const float*)d_in[1];
    const int n = in_sizes[0] / DIM;          // 500000
    float* out = (float*)d_out;

    // Workspace layout: chi[K*D] f16 | clo[K*D] f16 | csq[K] f32  (~65.5 KB)
    _Float16* chi = (_Float16*)d_ws;
    _Float16* clo = chi + NCLUST * DIM;
    float*    csq = (float*)(clo + NCLUST * DIM);

    // Output head = centroids passthrough (reference returns centroids[None,:,:]).
    hipMemcpyAsync(out, c, NCLUST * DIM * sizeof(float),
                   hipMemcpyDeviceToDevice, stream);

    kmeans_prep_centroids<<<1, THREADS, 0, stream>>>(c, chi, clo, csq);

    const int blocks = (n + PTS_PER_BLOCK - 1) / PTS_PER_BLOCK;
    kmeans_assign_wmma<<<blocks, THREADS, 0, stream>>>(
        x, chi, clo, csq, out + NCLUST * DIM, n);
}
